// GPTBlock_48258252538277
// MI455X (gfx1250) — matrix operand order repaired and checked by reading, not compiled, not run on hardware
//
#include <hip/hip_runtime.h>
#include <hip/hip_bf16.h>
#include <math.h>

typedef __bf16 bf16;
typedef __attribute__((ext_vector_type(8)))  bf16  v8bf;
typedef __attribute__((ext_vector_type(16))) bf16  v16bf;
typedef __attribute__((ext_vector_type(8)))  float v8f;

#define HID   2048
#define SEQ   2048
#define BATCH 2
#define NHEAD 16
#define HDIM  128
#define MROWS (BATCH*SEQ)   // 4096 rows in every [M,K] GEMM

static __device__ __forceinline__ v16bf cat16(v8bf lo, v8bf hi) {
  return __builtin_shufflevector(lo, hi, 0,1,2,3,4,5,6,7,8,9,10,11,12,13,14,15);
}
static __device__ __forceinline__ v8f wmma_bf(v16bf a, v16bf b, v8f c) {
  // D = A(16x32 bf16) x B(32x16 bf16) + C(16x16 f32)
  return __builtin_amdgcn_wmma_f32_16x16x32_bf16(false, a, false, b, (short)0, c, false, false);
}
static __device__ __forceinline__ float gelu_f(float x) {
  const float k = 0.7978845608028654f;
  return 0.5f * x * (1.0f + tanhf(k * (x + 0.044715f * x * x * x)));
}
// CDNA5 async copy: LDS[ldsaddr+off] = MEM[gaddr+off], 16B per lane, ASYNCcnt
static __device__ __forceinline__ void async_b128(unsigned ldsaddr, const void* gaddr) {
  asm volatile("global_load_async_to_lds_b128 %0, %1, off"
               :: "v"(ldsaddr), "v"(gaddr) : "memory");
}
static __device__ __forceinline__ void async_b128_o16(unsigned ldsaddr, const void* gaddr) {
  asm volatile("global_load_async_to_lds_b128 %0, %1, off offset:16"
               :: "v"(ldsaddr), "v"(gaddr) : "memory");
}
static __device__ __forceinline__ void wait_async0() {
  asm volatile("s_wait_asynccnt 0x0" ::: "memory");
}

// ---------------------------------------------------------------------------
// fp32 [K][N] -> bf16 W^T [N][K] (LDS tiled transpose, 32x32 tiles)
// ---------------------------------------------------------------------------
__global__ __launch_bounds__(256) void transpose_to_bf16(
    const float* __restrict__ w, bf16* __restrict__ wt, int K, int N) {
  __shared__ bf16 tile[32][33];
  const int n0 = blockIdx.x * 32;
  const int k0 = blockIdx.y * 32;
  const int tx = threadIdx.x & 31;
  const int ty = threadIdx.x >> 5;            // 0..7
  #pragma unroll
  for (int i = 0; i < 4; ++i) {
    const int k = k0 + ty + i * 8;
    tile[ty + i * 8][tx] = (bf16)w[(size_t)k * N + n0 + tx];
  }
  __syncthreads();
  #pragma unroll
  for (int i = 0; i < 4; ++i) {
    const int n = n0 + ty + i * 8;
    wt[(size_t)n * K + k0 + tx] = tile[tx][ty + i * 8];
  }
}

// ---------------------------------------------------------------------------
// LayerNorm over H=2048, one row per 256-thread block, bf16 output
// ---------------------------------------------------------------------------
__global__ __launch_bounds__(256) void ln_kernel(
    const float* __restrict__ x, const float* __restrict__ g,
    const float* __restrict__ b, bf16* __restrict__ out) {
  const int row = blockIdx.x;
  const float* xr = x + (size_t)row * HID;
  float v[8], lsum = 0.f, lsq = 0.f;
  #pragma unroll
  for (int i = 0; i < 8; ++i) {
    v[i] = xr[threadIdx.x + i * 256];
    lsum += v[i]; lsq += v[i] * v[i];
  }
  #pragma unroll
  for (int off = 16; off > 0; off >>= 1) {
    lsum += __shfl_down(lsum, off);
    lsq  += __shfl_down(lsq,  off);
  }
  __shared__ float s1[8], s2[8];
  const int w = threadIdx.x >> 5, l = threadIdx.x & 31;
  if (l == 0) { s1[w] = lsum; s2[w] = lsq; }
  __syncthreads();
  if (threadIdx.x == 0) {
    float a = 0.f, c = 0.f;
    for (int i = 0; i < 8; ++i) { a += s1[i]; c += s2[i]; }
    s1[0] = a; s2[0] = c;
  }
  __syncthreads();
  const float mean = s1[0] * (1.0f / HID);
  const float var  = s2[0] * (1.0f / HID) - mean * mean;
  const float rstd = rsqrtf(var + 1e-5f);
  #pragma unroll
  for (int i = 0; i < 8; ++i) {
    const int c = threadIdx.x + i * 256;
    out[(size_t)row * HID + c] = (bf16)((v[i] - mean) * rstd * g[c] + b[c]);
  }
}

// ---------------------------------------------------------------------------
// WMMA GEMM: C[M,N] = A[M,K](bf16) x W^T[N,K](bf16), + epilogue by MODE
//   MODE 0: +bias, scatter to Q[BH,S,HD], K[BH,S,HD], V^T[BH,HD,S] (bf16)
//   MODE 1: +bias +resid -> fp32 outf
//   MODE 2: gelu(+bias)  -> bf16 outb (row stride N)
//   MODE 3: +bias +resid -> fp32 outf (d_out)
// 256 threads = 8 waves; workgroup tile 128x128; wave tile 32x64; K step 32.
// Tiles staged with GLOBAL_LOAD_ASYNC_TO_LDS_B128 (ASYNCcnt), no VGPR bounce.
// ---------------------------------------------------------------------------
template<int MODE>
__global__ __launch_bounds__(256) void gemm_wmma(
    const bf16* __restrict__ A, const bf16* __restrict__ Wt,
    const float* __restrict__ bias, int K, int N,
    const float* __restrict__ resid, float* __restrict__ outf,
    bf16* __restrict__ outb,
    bf16* __restrict__ q, bf16* __restrict__ kk, bf16* __restrict__ vt) {
  __shared__ bf16 sA[128 * 40];   // 32-half rows padded to 40 (bank spread)
  __shared__ bf16 sB[128 * 40];
  const int t    = threadIdx.x;
  const int lane = t & 31;
  const int w    = t >> 5;
  const int wr   = w & 3;          // wave row  (4)
  const int wc   = w >> 2;         // wave col  (2)
  const int lm   = lane & 15;
  const int hi   = lane >> 4;      // half-wave select
  const int m0   = blockIdx.y * 128;
  const int n0   = blockIdx.x * 128;
  const int r    = t >> 1;         // cooperative-load tile row 0..127
  const int kc   = (t & 1) * 16;   // 16-half chunk within the 32-half row

  // LDS byte offsets for this thread's staging slots (flat addr low 32 bits)
  const unsigned ldsA = (unsigned)(size_t)(const void*)(sA + r * 40 + kc);
  const unsigned ldsB = (unsigned)(size_t)(const void*)(sB + r * 40 + kc);

  v8f acc[2][4] = {};
  for (int k0 = 0; k0 < K; k0 += 32) {
    const bf16* ga = A  + (size_t)(m0 + r) * K + k0 + kc;
    const bf16* gb = Wt + (size_t)(n0 + r) * K + k0 + kc;
    async_b128(ldsA, ga);          // 16B -> sA[r][kc..kc+7]
    async_b128_o16(ldsA, ga);      // 16B -> sA[r][kc+8..kc+15]
    async_b128(ldsB, gb);
    async_b128_o16(ldsB, gb);
    if (k0 + 32 < K) {             // global_prefetch next K slice
      __builtin_prefetch(ga + 32, 0, 0);
      __builtin_prefetch(gb + 32, 0, 0);
    }
    wait_async0();                 // own wave's async copies landed in LDS
    __syncthreads();               // all waves' copies visible

    const int kaA = hi ? 8  : 0;   // A-frag: lanes0-15 K{0..7,16..23}; 16-31 K{8..15,24..31}
    const int kaB = hi ? 8  : 0;   // B-frag: lanes0-15 K{0..7,16..23}; 16-31 K{8..15,24..31}
    v16bf af[2], bfr[4];
    #pragma unroll
    for (int mi = 0; mi < 2; ++mi) {
      const bf16* p = sA + (wr * 32 + mi * 16 + lm) * 40 + kaA;
      af[mi] = cat16(*(const v8bf*)p, *(const v8bf*)(p + 16));
    }
    #pragma unroll
    for (int ni = 0; ni < 4; ++ni) {
      const bf16* p = sB + (wc * 64 + ni * 16 + lm) * 40 + kaB;
      bfr[ni] = cat16(*(const v8bf*)p, *(const v8bf*)(p + 16));
    }
    #pragma unroll
    for (int mi = 0; mi < 2; ++mi)
      #pragma unroll
      for (int ni = 0; ni < 4; ++ni)
        acc[mi][ni] = wmma_bf(af[mi], bfr[ni], acc[mi][ni]);
    __syncthreads();
  }

  // ---- epilogue (C layout: lane col = lm; VGPR i row = i + 8*hi) ----
  #pragma unroll
  for (int mi = 0; mi < 2; ++mi) {
    #pragma unroll
    for (int ni = 0; ni < 4; ++ni) {
      const int col = n0 + wc * 64 + ni * 16 + lm;
      const float bv = bias[col];
      #pragma unroll
      for (int i = 0; i < 8; ++i) {
        const int row = m0 + wr * 32 + mi * 16 + i + hi * 8;
        const float v = acc[mi][ni][i] + bv;
        if (MODE == 0) {
          const int part   = col >> 11;       // 0:q 1:k 2:v
          const int within = col & 2047;
          const int head   = within >> 7;
          const int d      = within & 127;
          const int bb     = row >> 11;
          const int sidx   = row & 2047;
          const size_t bh  = (size_t)(bb * NHEAD + head);
          if (part == 0)      q [(bh * SEQ + sidx) * HDIM + d] = (bf16)v;
          else if (part == 1) kk[(bh * SEQ + sidx) * HDIM + d] = (bf16)v;
          else                vt[(bh * HDIM + d) * SEQ + sidx] = (bf16)v;
        } else if (MODE == 2) {
          outb[(size_t)row * N + col] = (bf16)gelu_f(v);
        } else { // MODE 1 / 3
          outf[(size_t)row * HID + col] = v + resid[(size_t)row * HID + col];
        }
      }
    }
  }
}

// ---------------------------------------------------------------------------
// Flash-style causal attention. Grid: (S/64, B*NH); 128 threads = 4 waves.
// Each wave owns 16 query rows; streams 32-key blocks; WMMA for QK^T and PV.
// ---------------------------------------------------------------------------
__global__ __launch_bounds__(128) void attn_kernel(
    const bf16* __restrict__ Q, const bf16* __restrict__ Kb,
    const bf16* __restrict__ Vt, bf16* __restrict__ O) {
  const int bh   = blockIdx.y;
  const int w    = threadIdx.x >> 5;
  const int lane = threadIdx.x & 31;
  const int lm   = lane & 15;
  const int hi   = lane >> 4;
  const int qr0  = blockIdx.x * 64 + w * 16;
  const bf16* qp = Q  + (size_t)bh * SEQ * HDIM;
  const bf16* kp = Kb + (size_t)bh * SEQ * HDIM;
  const bf16* vp = Vt + (size_t)bh * HDIM * SEQ;

  __shared__ bf16 sP[4 * 16 * 40];   // per-wave probs staging (C->A relayout)
  bf16* sp = sP + w * 16 * 40;

  const int kaA = hi ? 8 : 0;
  v16bf qf[4];                       // resident Q fragments across HD=128
  #pragma unroll
  for (int c = 0; c < 4; ++c) {
    const bf16* p = qp + (size_t)(qr0 + lm) * HDIM + c * 32 + kaA;
    qf[c] = cat16(*(const v8bf*)p, *(const v8bf*)(p + 16));
  }

  v8f acc[8] = {};
  float M[8], L[8];
  #pragma unroll
  for (int i = 0; i < 8; ++i) { M[i] = -1e30f; L[i] = 0.f; }

  const int nkb = (qr0 + 16 + 31) >> 5;     // causal #key-blocks of 32
  for (int kb = 0; kb < nkb; ++kb) {
    const int kbase = kb * 32;
    v8f s[2] = {};
    #pragma unroll
    for (int kt = 0; kt < 2; ++kt)
      #pragma unroll
      for (int c = 0; c < 4; ++c) {
        const bf16* p = kp + (size_t)(kbase + kt * 16 + lm) * HDIM
                           + c * 32 + (hi ? 8 : 0);
        s[kt] = wmma_bf(qf[c], cat16(*(const v8bf*)p, *(const v8bf*)(p + 16)), s[kt]);
      }

    const float sc = 0.08838834764831845f;  // 1/sqrt(128)
    #pragma unroll
    for (int i = 0; i < 8; ++i) {
      const int qrow = qr0 + i + hi * 8;
      float rmax = -1e30f;
      #pragma unroll
      for (int kt = 0; kt < 2; ++kt) {
        const int col = kbase + kt * 16 + lm;
        float v = s[kt][i] * sc;
        v = (col <= qrow) ? v : -1e30f;
        s[kt][i] = v;
        rmax = fmaxf(rmax, v);
      }
      #pragma unroll
      for (int off = 1; off < 16; off <<= 1)
        rmax = fmaxf(rmax, __shfl_xor(rmax, off));   // half-wave reduce
      const float mnew = fmaxf(M[i], rmax);
      const float corr = __expf(M[i] - mnew);
      const float p0 = __expf(s[0][i] - mnew);
      const float p1 = __expf(s[1][i] - mnew);
      sp[(i + hi * 8) * 40 + lm]      = (bf16)p0;    // C-layout -> LDS
      sp[(i + hi * 8) * 40 + 16 + lm] = (bf16)p1;
      float psum = p0 + p1;
      #pragma unroll
      for (int off = 1; off < 16; off <<= 1)
        psum += __shfl_xor(psum, off);
      L[i] = L[i] * corr + psum;
      M[i] = mnew;
      #pragma unroll
      for (int nt = 0; nt < 8; ++nt) acc[nt][i] *= corr;
    }

    // reload probs in A-fragment layout (same-wave DS ops are in-order)
    const bf16* pr = sp + lm * 40 + kaA;
    const v16bf pf = cat16(*(const v8bf*)pr, *(const v8bf*)(pr + 16));
    #pragma unroll
    for (int nt = 0; nt < 8; ++nt) {
      const bf16* p = vp + (size_t)(nt * 16 + lm) * SEQ + kbase + (hi ? 8 : 0);
      acc[nt] = wmma_bf(pf, cat16(*(const v8bf*)p, *(const v8bf*)(p + 16)), acc[nt]);
    }
  }

  const int bb = bh >> 4, h = bh & 15;
  #pragma unroll
  for (int nt = 0; nt < 8; ++nt) {
    const int col = h * HDIM + nt * 16 + lm;
    #pragma unroll
    for (int i = 0; i < 8; ++i) {
      const int srow = qr0 + i + hi * 8;
      O[((size_t)bb * SEQ + srow) * HID + col] = (bf16)(acc[nt][i] / L[i]);
    }
  }
}

// ---------------------------------------------------------------------------
extern "C" void kernel_launch(void* const* d_in, const int* in_sizes, int n_in,
                              void* d_out, int out_size, void* d_ws, size_t ws_size,
                              hipStream_t stream) {
  (void)in_sizes; (void)n_in; (void)out_size; (void)ws_size;
  const float* x      = (const float*)d_in[0];
  const float* ln1_g  = (const float*)d_in[1];
  const float* ln1_b  = (const float*)d_in[2];
  const float* ln2_g  = (const float*)d_in[3];
  const float* ln2_b  = (const float*)d_in[4];
  const float* w_qkv  = (const float*)d_in[5];
  const float* b_qkv  = (const float*)d_in[6];
  const float* w_attn = (const float*)d_in[7];
  const float* b_attn = (const float*)d_in[8];
  const float* w_ff1  = (const float*)d_in[9];
  const float* b_ff1  = (const float*)d_in[10];
  const float* w_ff2  = (const float*)d_in[11];
  const float* b_ff2  = (const float*)d_in[12];

  char* ws = (char*)d_ws;
  bf16* wqkv_t = (bf16*)(ws + 0);            // 6144x2048 bf16   (25 MB)
  bf16* watt_t = (bf16*)(ws + 25165824);     // 2048x2048 bf16   (8.4 MB)
  bf16* wff1_t = (bf16*)(ws + 33554432);     // 8192x2048 bf16   (33.6 MB)
  bf16* wff2_t = (bf16*)(ws + 67108864);     // 2048x8192 bf16   (33.6 MB)
  bf16* hln    = (bf16*)(ws + 100663296);    // 4096x2048 bf16   (ln1 & ln2)
  bf16* qb     = (bf16*)(ws + 117440512);    // [BH,S,HD]
  bf16* kbuf   = (bf16*)(ws + 134217728);    // [BH,S,HD]
  bf16* vtb    = (bf16*)(ws + 150994944);    // [BH,HD,S]
  bf16* attn   = (bf16*)(ws + 167772160);    // 4096x2048 bf16
  float* x1    = (float*)(ws + 184549376);   // 4096x2048 f32
  bf16* a1     = (bf16*)(ws + 218103808);    // 4096x8192 bf16
  float* out   = (float*)d_out;

  // 1) weights fp32 -> bf16 transposed [N][K]
  transpose_to_bf16<<<dim3(192, 64), 256, 0, stream>>>(w_qkv,  wqkv_t, 2048, 6144);
  transpose_to_bf16<<<dim3(64,  64), 256, 0, stream>>>(w_attn, watt_t, 2048, 2048);
  transpose_to_bf16<<<dim3(256, 64), 256, 0, stream>>>(w_ff1,  wff1_t, 2048, 8192);
  transpose_to_bf16<<<dim3(64, 256), 256, 0, stream>>>(w_ff2,  wff2_t, 8192, 2048);

  // 2) LN1
  ln_kernel<<<MROWS, 256, 0, stream>>>(x, ln1_g, ln1_b, hln);
  // 3) QKV projection, scatter Q/K/V^T
  gemm_wmma<0><<<dim3(48, 32), 256, 0, stream>>>(hln, wqkv_t, b_qkv, 2048, 6144,
      nullptr, nullptr, nullptr, qb, kbuf, vtb);
  // 4) causal flash attention
  attn_kernel<<<dim3(32, 32), 128, 0, stream>>>(qb, kbuf, vtb, attn);
  // 5) out-projection + residual
  gemm_wmma<1><<<dim3(16, 32), 256, 0, stream>>>(attn, watt_t, b_attn, 2048, 2048,
      x, x1, nullptr, nullptr, nullptr, nullptr);
  // 6) LN2
  ln_kernel<<<MROWS, 256, 0, stream>>>(x1, ln2_g, ln2_b, hln);
  // 7) FF1 + gelu
  gemm_wmma<2><<<dim3(64, 32), 256, 0, stream>>>(hln, wff1_t, b_ff1, 2048, 8192,
      nullptr, nullptr, a1, nullptr, nullptr, nullptr);
  // 8) FF2 + residual -> d_out
  gemm_wmma<3><<<dim3(16, 32), 256, 0, stream>>>(a1, wff2_t, b_ff2, 8192, 2048,
      x1, out, nullptr, nullptr, nullptr, nullptr);
}
